// LayerGCN_40681930228297
// MI455X (gfx1250) — matrix operand order repaired and checked by reading, not compiled, not run on hardware
//
#include <hip/hip_runtime.h>
#include <hip/hip_bf16.h>
#include <math.h>

typedef __attribute__((ext_vector_type(16))) _Float16 v16h;
typedef __attribute__((ext_vector_type(8)))  float    v8f;

#define NUM_USER 100000
#define NUM_ITEM 50000
#define NTOT     150000
#define NEDGE    1200000
#define DDIM     64
#define FITEM    384
#define NLAYERS  4

// ---------------------------------------------------------------------------
// Hardware tanh: CDNA5 has V_TANH_F32 as a 1-op transcendental. Prefer the
// clang builtin when available; otherwise inline asm (v_nop satisfies the
// TRANS-op result hazard rule since the compiler can't see through the asm).
// ---------------------------------------------------------------------------
__device__ __forceinline__ float fast_tanh(float v) {
#if __has_builtin(__builtin_amdgcn_tanhf)
    return __builtin_amdgcn_tanhf(v);
#else
    float r;
    asm volatile("v_tanh_f32 %0, %1\n\tv_nop" : "=v"(r) : "v"(v));
    return r;
#endif
}

// ---------------------------------------------------------------------------
// Kernel 1: user ego rows: ego = user_fea + sum_k g_emb_u[k]; also init x, acc
// ---------------------------------------------------------------------------
__global__ __launch_bounds__(256) void k_user_ego(const float* __restrict__ user_fea,
                                                  const float* __restrict__ g_emb_u,
                                                  float* __restrict__ ego,
                                                  float* __restrict__ x,
                                                  float* __restrict__ acc) {
    long i = (long)blockIdx.x * blockDim.x + threadIdx.x;   // per float4
    const long total4 = (long)NUM_USER * DDIM / 4;
    if (i >= total4) return;
    int d4 = (int)(i & (DDIM / 4 - 1));                     // 0..15
    const float4* uf = (const float4*)user_fea;
    const float4* g  = (const float4*)g_emb_u;
    float4 v  = uf[i];
    float4 g0 = g[0 * (DDIM / 4) + d4];
    float4 g1 = g[1 * (DDIM / 4) + d4];
    float4 g2 = g[2 * (DDIM / 4) + d4];
    v.x += g0.x + g1.x + g2.x;
    v.y += g0.y + g1.y + g2.y;
    v.z += g0.z + g1.z + g2.z;
    v.w += g0.w + g1.w + g2.w;
    ((float4*)ego)[i] = v;
    ((float4*)x)[i]   = v;
    ((float4*)acc)[i] = v;
}

// ---------------------------------------------------------------------------
// Kernel 2: item ego rows via WMMA GEMM:
//   ego[U+m][n] = tanh(item_fea[m,:] . mlp_w[n,:] + b[n]) + sum_k g_emb_i[k][n]
// One wave32 per 16x16 output tile; K=384 in 12 steps of 32 via
// v_wmma_f32_16x16x32_f16 (f32 accumulation).
// ---------------------------------------------------------------------------
__global__ __launch_bounds__(256) void k_item_ego(const float* __restrict__ item_fea,
                                                  const float* __restrict__ mlp_w,
                                                  const float* __restrict__ mlp_b,
                                                  const float* __restrict__ g_emb_i,
                                                  float* __restrict__ ego,
                                                  float* __restrict__ x,
                                                  float* __restrict__ acc) {
    const int lane = threadIdx.x & 31;
    const int wave = threadIdx.x >> 5;
    const int tile = blockIdx.x * 8 + wave;                 // 8 waves / block
    const int nTilesN    = DDIM / 16;                       // 4
    const int totalTiles = (NUM_ITEM / 16) * nTilesN;       // 12500
    if (tile >= totalTiles) return;                         // wave-uniform exit

    const int mTile = tile / nTilesN;
    const int nTile = tile % nTilesN;
    const int m0 = mTile * 16;
    const int n0 = nTile * 16;
    const int l15 = lane & 15;
    const int hi  = lane >> 4;                              // 0 or 1

    const float* aRow = item_fea + (long)(m0 + l15) * FITEM;  // A row (M = lane&15)
    const float* bRow = mlp_w    + (long)(n0 + l15) * FITEM;  // B col n -> mlp_w row n

    v8f c = {};
    for (int k0 = 0; k0 < FITEM; k0 += 32) {
        v16h a, b;
        // A 16x32 f16 layout: halves 0..15 -> K = k0 + 16*hi + h, M = lane&15
        const int ka = k0 + hi * 16;
#pragma unroll
        for (int h = 0; h < 8; ++h) {
            a[h]     = (_Float16)aRow[ka + h];
            a[h + 8] = (_Float16)aRow[ka + 8 + h];
        }
        // B 32x16 f16 layout: halves 0..15 -> K = k0 + 16*hi + h, N = lane&15
        const int kb = k0 + hi * 16;
#pragma unroll
        for (int h = 0; h < 16; ++h) {
            b[h] = (_Float16)bRow[kb + h];
        }
        c = __builtin_amdgcn_wmma_f32_16x16x32_f16(
                /*neg_a=*/false, a, /*neg_b=*/false, b,
                /*c_mod=*/(short)0, c, /*reuse_a=*/false, /*reuse_b=*/false);
    }

    const int nCol = n0 + l15;
    const float bias = mlp_b[nCol];
    const float gsum = g_emb_i[0 * DDIM + nCol] + g_emb_i[1 * DDIM + nCol] +
                       g_emb_i[2 * DDIM + nCol];
#pragma unroll
    for (int r = 0; r < 8; ++r) {
        // C/D layout: VGPR r -> M = r (lanes 0-15) / r+8 (lanes 16-31), N = lane&15
        int row = m0 + r + hi * 8;
        float v = fast_tanh(c[r] + bias) + gsum;
        long off = (long)(NUM_USER + row) * DDIM + nCol;
        ego[off] = v;
        x[off]   = v;
        acc[off] = v;
    }
}

// ---------------------------------------------------------------------------
// Kernel 3: zero x_new (float4 stores)
// ---------------------------------------------------------------------------
__global__ __launch_bounds__(256) void k_zero(float4* __restrict__ p, long n4) {
    long i = (long)blockIdx.x * blockDim.x + threadIdx.x;
    if (i >= n4) return;
    float4 z = {0.f, 0.f, 0.f, 0.f};
    p[i] = z;
}

// ---------------------------------------------------------------------------
// Kernel 4: edge-parallel SpMM scatter: x_new[row] += val * x[col]
// 8 threads per edge, 8 floats each -> hardware fp32 atomics.
// ---------------------------------------------------------------------------
__global__ __launch_bounds__(256) void k_spmm(const int* __restrict__ adj_row,
                                              const int* __restrict__ adj_col,
                                              const float* __restrict__ adj_val,
                                              const float* __restrict__ xin,
                                              float* __restrict__ xout) {
    long gid = (long)blockIdx.x * blockDim.x + threadIdx.x;
    const long nwork = (long)NEDGE * 8;
    if (gid >= nwork) return;
    int e = (int)(gid >> 3);
    int g = (int)(gid & 7);
    int r = adj_row[e];
    int cidx = adj_col[e];
    float v = adj_val[e];
    const float* src = xin  + (long)cidx * DDIM + g * 8;
    float* dst       = xout + (long)r    * DDIM + g * 8;
    float4 s0 = ((const float4*)src)[0];
    float4 s1 = ((const float4*)src)[1];
    unsafeAtomicAdd(&dst[0], v * s0.x);
    unsafeAtomicAdd(&dst[1], v * s0.y);
    unsafeAtomicAdd(&dst[2], v * s0.z);
    unsafeAtomicAdd(&dst[3], v * s0.w);
    unsafeAtomicAdd(&dst[4], v * s1.x);
    unsafeAtomicAdd(&dst[5], v * s1.y);
    unsafeAtomicAdd(&dst[6], v * s1.z);
    unsafeAtomicAdd(&dst[7], v * s1.w);
}

// ---------------------------------------------------------------------------
// Kernel 5: cosine re-weight + accumulate. One wave32 per row; lane l owns
// columns {2l, 2l+1} so every access is a contiguous b64 load/store.
//   w = dot(x_new,ego) / (max(|x_new|,eps)*max(|ego|,eps))
//   x = w * x_new ; acc += x
// ---------------------------------------------------------------------------
__global__ __launch_bounds__(256) void k_cosacc(const float* __restrict__ xnew,
                                                const float* __restrict__ ego,
                                                float* __restrict__ x,
                                                float* __restrict__ acc) {
    const int lane = threadIdx.x & 31;
    const int wave = threadIdx.x >> 5;
    long row = (long)blockIdx.x * 8 + wave;
    if (row >= NTOT) return;
    const float2* xr = (const float2*)(xnew + row * DDIM);
    const float2* er = (const float2*)(ego  + row * DDIM);
    float2 a = xr[lane];
    float2 e = er[lane];
    float dot = a.x * e.x + a.y * e.y;
    float nx  = a.x * a.x + a.y * a.y;
    float ne  = e.x * e.x + e.y * e.y;
#pragma unroll
    for (int s = 16; s > 0; s >>= 1) {
        dot += __shfl_xor(dot, s, 32);
        nx  += __shfl_xor(nx,  s, 32);
        ne  += __shfl_xor(ne,  s, 32);
    }
    float w = dot / (fmaxf(sqrtf(nx), 1e-8f) * fmaxf(sqrtf(ne), 1e-8f));
    float2 sv = {w * a.x, w * a.y};
    float2* xo = (float2*)(x + row * DDIM);
    float2* ao = (float2*)(acc + row * DDIM);
    xo[lane] = sv;
    float2 av = ao[lane];
    av.x += sv.x;
    av.y += sv.y;
    ao[lane] = av;
}

// ---------------------------------------------------------------------------
// Host-side orchestration (graph-capture safe: only kernel launches on stream)
// ---------------------------------------------------------------------------
extern "C" void kernel_launch(void* const* d_in, const int* in_sizes, int n_in,
                              void* d_out, int out_size, void* d_ws, size_t ws_size,
                              hipStream_t stream) {
    const int*   adj_row  = (const int*)d_in[0];
    const int*   adj_col  = (const int*)d_in[1];
    const float* adj_val  = (const float*)d_in[2];
    const float* user_fea = (const float*)d_in[3];
    const float* item_fea = (const float*)d_in[4];
    const float* g_emb_u  = (const float*)d_in[5];
    const float* g_emb_i  = (const float*)d_in[6];
    const float* mlp_w    = (const float*)d_in[7];
    const float* mlp_b    = (const float*)d_in[8];

    float* acc = (float*)d_out;                          // [NTOT, 64]

    const long rowFloats = (long)NTOT * DDIM;            // 9.6M floats
    float* ego  = (float*)d_ws;
    float* x    = ego + rowFloats;
    float* xnew = x + rowFloats;

    // Setup: ego / x / acc
    {
        long total4 = (long)NUM_USER * DDIM / 4;
        int blocks = (int)((total4 + 255) / 256);
        k_user_ego<<<blocks, 256, 0, stream>>>(user_fea, g_emb_u, ego, x, acc);
    }
    {
        int totalTiles = (NUM_ITEM / 16) * (DDIM / 16);  // 12500 wave-tiles
        int blocks = (totalTiles + 7) / 8;               // 8 waves per 256-thread block
        k_item_ego<<<blocks, 256, 0, stream>>>(item_fea, mlp_w, mlp_b, g_emb_i,
                                               ego, x, acc);
    }

    // 4 propagation layers
    for (int layer = 0; layer < NLAYERS; ++layer) {
        {
            long n4 = rowFloats / 4;
            int blocks = (int)((n4 + 255) / 256);
            k_zero<<<blocks, 256, 0, stream>>>((float4*)xnew, n4);
        }
        {
            long nwork = (long)NEDGE * 8;
            int blocks = (int)((nwork + 255) / 256);
            k_spmm<<<blocks, 256, 0, stream>>>(adj_row, adj_col, adj_val, x, xnew);
        }
        {
            int blocks = (NTOT + 7) / 8;
            k_cosacc<<<blocks, 256, 0, stream>>>(xnew, ego, x, acc);
        }
    }
}
